// SelfDWConv2d_5454608465982
// MI455X (gfx1250) — compile-verified, matrix operand order checked by reading, not run on hardware
//
#include <hip/hip_runtime.h>

// ---------------- problem constants ----------------
#define BB   32
#define CC   256
#define HH   56
#define WW   56
#define HWV  3136        // HH*WW
#define NKV  16
#define KSV  7
#define KKV  49          // KSV*KSV
#define TEMP 512.0f      // 2*C
#define EPSV 1e-5f

// ---------------- vector types ----------------
typedef __attribute__((ext_vector_type(16))) __bf16 v16bf;
typedef __attribute__((ext_vector_type(8)))  __bf16 v8bf;
typedef __attribute__((ext_vector_type(4)))  __bf16 v4bf;
typedef __attribute__((ext_vector_type(8)))  float  v8f;

union bfrag { v16bf v; v8bf h[2]; };

// ---------------- workspace layout (bytes) ----------------
// xn  : B*C*HW bf16  = 51,380,224
// G   : B*C*C  f32   =  8,388,608
// T   : B*C*NK f32   =    524,288
// wgt : B*C*49 f32   =  1,605,632
#define WS_XN   0
#define WS_G    (51380224)
#define WS_T    (WS_G + 8388608)
#define WS_WGT  (WS_T + 524288)

// =====================================================================
// Kernel 1: BatchNorm (eval) -> bf16, vectorized x4
// =====================================================================
__global__ __launch_bounds__(256) void bn_bf16_kernel(
    const float* __restrict__ x,
    const float* __restrict__ gamma, const float* __restrict__ beta,
    const float* __restrict__ mean,  const float* __restrict__ var,
    __bf16* __restrict__ xn, int total4)
{
    int i4 = blockIdx.x * blockDim.x + threadIdx.x;
    if (i4 >= total4) return;
    int c = (i4 / (HWV / 4)) % CC;           // HW divisible by 4 -> channel const per float4
    float s = gamma[c] * rsqrtf(var[c] + EPSV);
    float t = beta[c] - mean[c] * s;
    float4 v = ((const float4*)x)[i4];
    v4bf o;
    o.x = (__bf16)(v.x * s + t);
    o.y = (__bf16)(v.y * s + t);
    o.z = (__bf16)(v.z * s + t);
    o.w = (__bf16)(v.w * s + t);
    ((v4bf*)xn)[i4] = o;
}

// =====================================================================
// Kernel 2: per-batch Gram matrix G[b] = xn[b] (CxHW) * xn[b]^T  via WMMA bf16
// One wave computes a 64x64 tile as 4x4 grid of 16x16 WMMA tiles.
// grid = B*16 waves / 8 waves-per-block = 64 blocks of 256 threads.
// =====================================================================
__global__ __launch_bounds__(256) void gram_wmma_kernel(
    const __bf16* __restrict__ xn, float* __restrict__ G)
{
    const int lane  = threadIdx.x & 31;
    const int wave  = threadIdx.x >> 5;
    const int gwave = blockIdx.x * 8 + wave;     // 0 .. 511
    const int b     = gwave >> 4;                // batch
    const int tile  = gwave & 15;                // 4x4 macro-tiles of 64
    const int m0    = (tile >> 2) * 64;
    const int n0    = (tile &  3) * 64;

    const __bf16* Xb = xn + (size_t)b * CC * HWV;

    const int lrow = lane & 15;        // row within 16-tile held by this lane
    const int kofs = (lane >> 4) * 8;  // K sub-chunk select (ISA 16-bit A layout)

    const __bf16* pA[4];
    const __bf16* pB[4];
#pragma unroll
    for (int i = 0; i < 4; ++i) {
        pA[i] = Xb + (size_t)(m0 + i * 16 + lrow) * HWV + kofs;
        pB[i] = Xb + (size_t)(n0 + i * 16 + lrow) * HWV + kofs;
    }

    v8f acc[4][4];
    const v8f vzero = {0.f,0.f,0.f,0.f,0.f,0.f,0.f,0.f};
#pragma unroll
    for (int i = 0; i < 4; ++i)
#pragma unroll
        for (int j = 0; j < 4; ++j) acc[i][j] = vzero;

    for (int k = 0; k < HWV; k += 32) {
        bfrag a[4], bf[4];
#pragma unroll
        for (int i = 0; i < 4; ++i) {
            a[i].h[0]  = *(const v8bf*)(pA[i] + k);
            a[i].h[1]  = *(const v8bf*)(pA[i] + k + 16);
            bf[i].h[0] = *(const v8bf*)(pB[i] + k);
            bf[i].h[1] = *(const v8bf*)(pB[i] + k + 16);
        }
        if (k + 128 < HWV) {            // gfx1250 global_prefetch_b8
            __builtin_prefetch(pA[0] + k + 128, 0, 1);
            __builtin_prefetch(pB[0] + k + 128, 0, 1);
        }
#pragma unroll
        for (int i = 0; i < 4; ++i)
#pragma unroll
            for (int j = 0; j < 4; ++j)
                acc[i][j] = __builtin_amdgcn_wmma_f32_16x16x32_bf16(
                    false, a[i].v, false, bf[j].v,
                    (short)0, acc[i][j], false, false);
    }

    // Store: C/D layout -> VGPR r: lanes 0-15 = (M=r, N=lane), lanes 16-31 = (M=r+8, N=lane-16)
    const int rhalf = (lane >> 4) * 8;
    float* Gb = G + (size_t)b * CC * CC;
#pragma unroll
    for (int i = 0; i < 4; ++i)
#pragma unroll
        for (int j = 0; j < 4; ++j) {
            int col = n0 + j * 16 + lrow;
#pragma unroll
            for (int r = 0; r < 8; ++r) {
                int row = m0 + i * 16 + r + rhalf;
                Gb[(size_t)row * CC + col] = acc[i][j][r];
            }
        }
}

// =====================================================================
// Kernel 3a: T[b,j,k] = sum_i G[b,j,i] * k_w[k,i]      (B*C*NK threads)
// =====================================================================
__global__ __launch_bounds__(256) void gkw_kernel(
    const float* __restrict__ G, const float* __restrict__ k_w,
    float* __restrict__ T)
{
    int t = blockIdx.x * blockDim.x + threadIdx.x;   // B*C*NK = 131072
    int k = t & 15;
    int j = (t >> 4) & 255;
    int b = t >> 12;
    const float* Gr = G + ((size_t)b * CC + j) * CC;
    const float* kw = k_w + (size_t)k * CC;
    float s = 0.f;
#pragma unroll 8
    for (int i = 0; i < CC; ++i) s += Gr[i] * kw[i];
    T[t] = s;
}

// =====================================================================
// Kernel 3b: logits = q_w * T ; softmax over NK ; wgt = attn * bank
// one thread per (b,c) -> 8192 threads
// =====================================================================
__global__ __launch_bounds__(256) void attn_wgt_kernel(
    const float* __restrict__ T, const float* __restrict__ q_w,
    const float* __restrict__ bank, float* __restrict__ wgt)
{
    int t = blockIdx.x * blockDim.x + threadIdx.x;   // B*C = 8192
    int c = t & 255;
    int b = t >> 8;
    const float* q  = q_w + (size_t)c * CC;
    const float* Tb = T + (size_t)b * CC * NKV;

    float acc[NKV];
#pragma unroll
    for (int k = 0; k < NKV; ++k) acc[k] = 0.f;
    for (int j = 0; j < CC; ++j) {
        float qv = q[j];
        const float4* Tr = (const float4*)(Tb + (size_t)j * NKV);
#pragma unroll
        for (int k4 = 0; k4 < 4; ++k4) {
            float4 v = Tr[k4];
            acc[k4*4+0] += qv * v.x;
            acc[k4*4+1] += qv * v.y;
            acc[k4*4+2] += qv * v.z;
            acc[k4*4+3] += qv * v.w;
        }
    }
    // softmax over NK with /temp
    float m = -3.402823466e+38f;
#pragma unroll
    for (int k = 0; k < NKV; ++k) { acc[k] *= (1.0f / TEMP); m = fmaxf(m, acc[k]); }
    float sum = 0.f;
#pragma unroll
    for (int k = 0; k < NKV; ++k) { acc[k] = __expf(acc[k] - m); sum += acc[k]; }
    float inv = 1.0f / sum;
#pragma unroll
    for (int k = 0; k < NKV; ++k) acc[k] *= inv;

    // wgt[b,c,:] = attn . bank   (NK x 49)
    float* wp = wgt + (size_t)t * KKV;
    for (int e = 0; e < KKV; ++e) {
        float s = 0.f;
#pragma unroll
        for (int k = 0; k < NKV; ++k) s += acc[k] * bank[k * KKV + e];
        wp[e] = s;
    }
}

// =====================================================================
// Kernel 4: dynamic depthwise 7x7 conv (same pad) + bias, LDS halo tile
// one block per (b,c) plane -> 8192 blocks
// =====================================================================
__global__ __launch_bounds__(256) void dwconv_kernel(
    const float* __restrict__ x, const float* __restrict__ wgt,
    const float* __restrict__ bias, float* __restrict__ out)
{
    __shared__ float tile[62 * 62];
    __shared__ float kw[KKV];

    const int bc = blockIdx.x;           // b*C + c
    const int c  = bc & 255;
    const float* xp = x + (size_t)bc * HWV;

    for (int i = threadIdx.x; i < 62 * 62; i += blockDim.x) {
        int ty = i / 62, tx = i - ty * 62;
        int sy = ty - 3, sx = tx - 3;
        float v = 0.f;
        if (sy >= 0 && sy < HH && sx >= 0 && sx < WW) v = xp[sy * WW + sx];
        tile[i] = v;
    }
    if (threadIdx.x < KKV) kw[threadIdx.x] = wgt[(size_t)bc * KKV + threadIdx.x];
    __syncthreads();

    const float bv = bias[c];
    float* op = out + (size_t)bc * HWV;
    for (int i = threadIdx.x; i < HWV; i += blockDim.x) {
        int y = i / WW, xx = i - y * WW;
        float s = bv;
#pragma unroll
        for (int dy = 0; dy < KSV; ++dy) {
            const float* tr = &tile[(y + dy) * 62 + xx];
#pragma unroll
            for (int dx = 0; dx < KSV; ++dx)
                s += tr[dx] * kw[dy * KSV + dx];
        }
        op[i] = s;
    }
}

// =====================================================================
// launch
// =====================================================================
extern "C" void kernel_launch(void* const* d_in, const int* in_sizes, int n_in,
                              void* d_out, int out_size, void* d_ws, size_t ws_size,
                              hipStream_t stream) {
    const float* x     = (const float*)d_in[0];
    const float* gamma = (const float*)d_in[1];
    const float* beta  = (const float*)d_in[2];
    const float* mean  = (const float*)d_in[3];
    const float* var   = (const float*)d_in[4];
    const float* q_w   = (const float*)d_in[5];
    const float* k_w   = (const float*)d_in[6];
    const float* bank  = (const float*)d_in[7];
    const float* bias  = (const float*)d_in[8];
    float* out = (float*)d_out;

    char* ws = (char*)d_ws;
    __bf16* xn = (__bf16*)(ws + WS_XN);
    float*  G  = (float*)(ws + WS_G);
    float*  T  = (float*)(ws + WS_T);
    float*  Wg = (float*)(ws + WS_WGT);

    // 1) BN -> bf16 (x4 vectorized): 6,422,528 quads
    {
        int total4 = BB * CC * HWV / 4;
        int blocks = (total4 + 255) / 256;
        bn_bf16_kernel<<<blocks, 256, 0, stream>>>(x, gamma, beta, mean, var, xn, total4);
    }
    // 2) Gram via WMMA: 512 waves -> 64 blocks
    gram_wmma_kernel<<<64, 256, 0, stream>>>(xn, G);
    // 3a) T = G * k_w^T : 131072 threads
    gkw_kernel<<<(BB * CC * NKV) / 256, 256, 0, stream>>>(G, k_w, T);
    // 3b) logits/softmax/weights : 8192 threads
    attn_wgt_kernel<<<(BB * CC) / 256, 256, 0, stream>>>(T, q_w, bank, Wg);
    // 4) depthwise conv + bias : one block per (b,c)
    dwconv_kernel<<<BB * CC, 256, 0, stream>>>(x, Wg, bias, out);
}
